// BaseMoE_25340307047014
// MI455X (gfx1250) — compile-verified, matrix operand order checked by reading, not run on hardware
//
#include <hip/hip_runtime.h>
#include <cstdint>
#include <cstddef>

// Problem constants (match reference)
#define N_TOK 8192
#define DIM   1024
#define HID   4096
#define NE    8

// Fused-kernel tiling
#define TM 32          // tokens per block
#define HC 64          // H-chunk per GEMM1->GEMM2 round
#define XS 1032        // LDS stride for x tile (1024 + 8 pad -> conflict-free frag reads)
#define HS 72          // LDS stride for h tile (64 + 8 pad)

typedef __attribute__((ext_vector_type(16))) __bf16 bf16x16;
typedef __attribute__((ext_vector_type(8)))  __bf16 bf16x8;
typedef __attribute__((ext_vector_type(4)))  __bf16 bf16x4;
typedef __attribute__((ext_vector_type(8)))  float  f32x8;
typedef __attribute__((ext_vector_type(4)))  float  f32x4;
typedef __attribute__((ext_vector_type(4)))  int    i32x4;

#define GLOBAL_AS __attribute__((address_space(1)))
#define LDS_AS    __attribute__((address_space(3)))

static __device__ __forceinline__ bf16x16 cat8(bf16x8 a, bf16x8 b) {
  return __builtin_shufflevector(a, b, 0,1,2,3,4,5,6,7,8,9,10,11,12,13,14,15);
}
static __device__ __forceinline__ f32x8 wmma_bf16(bf16x16 a, bf16x16 b, f32x8 c) {
  return __builtin_amdgcn_wmma_f32_16x16x32_bf16(false, a, false, b, (short)0, c,
                                                 false, false);
}

// ---------------------------------------------------------------- x -> bf16
__global__ __launch_bounds__(256) void cvt_x_kernel(const float* __restrict__ x,
                                                    __bf16* __restrict__ xb) {
  size_t i = ((size_t)blockIdx.x * 256 + threadIdx.x) * 4;
  f32x4 v = *(const f32x4*)(x + i);
  bf16x4 o;
  o.x = (__bf16)v.x; o.y = (__bf16)v.y; o.z = (__bf16)v.z; o.w = (__bf16)v.w;
  *(bf16x4*)(xb + i) = o;
}

// ----------------------------------------------- [E][R][C] f32 -> [E][C][R] bf16
__global__ __launch_bounds__(256) void transpose_cvt_kernel(
    const float* __restrict__ src, __bf16* __restrict__ dst, int R, int C) {
  __shared__ float tile[32][33];
  const int e  = blockIdx.z;
  const int c0 = blockIdx.x * 32;
  const int r0 = blockIdx.y * 32;
  const float* s = src + (size_t)e * R * C;
  __bf16*      d = dst + (size_t)e * R * C;
  const int tx = threadIdx.x, ty = threadIdx.y;
#pragma unroll
  for (int i = 0; i < 4; ++i) {
    int rr = ty + i * 8;
    tile[rr][tx] = s[(size_t)(r0 + rr) * C + c0 + tx];
  }
  __syncthreads();
#pragma unroll
  for (int i = 0; i < 4; ++i) {
    int cc = ty + i * 8;
    d[(size_t)(c0 + cc) * R + r0 + tx] = (__bf16)tile[tx][cc];
  }
}

// ------------------------------------------------- softmax router (full f32)
__global__ __launch_bounds__(256) void gate_kernel(const float* __restrict__ x,
                                                   const float* __restrict__ Wg,
                                                   const float* __restrict__ bg,
                                                   float* __restrict__ gw) {
  __shared__ float red[256][NE];
  const int n = blockIdx.x, t = threadIdx.x;
  float acc[NE];
#pragma unroll
  for (int e = 0; e < NE; ++e) acc[e] = 0.f;
  for (int it = 0; it < DIM / 256; ++it) {
    int dd = t + it * 256;
    float xv = x[(size_t)n * DIM + dd];
    const float* wr = Wg + (size_t)dd * NE;
#pragma unroll
    for (int e = 0; e < NE; ++e) acc[e] += xv * wr[e];
  }
#pragma unroll
  for (int e = 0; e < NE; ++e) red[t][e] = acc[e];
  __syncthreads();
  for (int s2 = 128; s2 > 0; s2 >>= 1) {
    if (t < s2) {
#pragma unroll
      for (int e = 0; e < NE; ++e) red[t][e] += red[t + s2][e];
    }
    __syncthreads();
  }
  if (t == 0) {
    float lg[NE], mx = -1e30f;
#pragma unroll
    for (int e = 0; e < NE; ++e) { lg[e] = red[0][e] + bg[e]; mx = fmaxf(mx, lg[e]); }
    float sum = 0.f;
#pragma unroll
    for (int e = 0; e < NE; ++e) { lg[e] = expf(lg[e] - mx); sum += lg[e]; }
    float inv = 1.f / sum;
#pragma unroll
    for (int e = 0; e < NE; ++e) gw[(size_t)n * NE + e] = lg[e] * inv;
  }
}

// --------------------------------------------------------- fused MoE kernel
// xb  : [N][D] bf16          (L2-resident after cvt)
// w1t : [E][H][D] bf16       (W1 transposed; B^T so frags are contiguous-K)
// w2t : [E][D][H] bf16       (W2 transposed)
// gw  : [N][E] f32 gate weights
__global__ __launch_bounds__(256) void moe_fused_kernel(
    const __bf16* __restrict__ xb, const __bf16* __restrict__ w1t,
    const __bf16* __restrict__ w2t, const float* __restrict__ gw,
    const float* __restrict__ b1g, const float* __restrict__ b2g,
    float* __restrict__ out) {
  __shared__ __align__(16) __bf16 xb_lds[TM * XS];       // 66,048 B
  __shared__ __align__(16) __bf16 h_lds[2][TM * HS];     //  9,216 B (ping-pong)
  __shared__ float wg_lds[TM][NE];                       //  1,024 B

  const int tid  = threadIdx.x;
  const int wave = tid >> 5;
  const int lane = tid & 31;
  const int lmod = lane & 15;
  const int lhalf = lane >> 4;
  const int tokenBase = blockIdx.x * TM;

  // stage gate weights (exactly 256 = TM*NE values, one per thread)
  wg_lds[tid >> 3][tid & 7] = gw[(size_t)(tokenBase + (tid >> 3)) * NE + (tid & 7)];

  // stage x tile 32x1024 bf16 (64 KB) with padded stride
#if __has_builtin(__builtin_amdgcn_global_load_async_to_lds_b128) && \
    __has_builtin(__builtin_amdgcn_s_wait_asynccnt)
  for (int i = 0; i < 16; ++i) {
    int c   = i * 256 + tid;
    int row = c >> 7;
    int col = (c & 127) * 8;
    __builtin_amdgcn_global_load_async_to_lds_b128(
        (GLOBAL_AS i32x4*)(xb + (size_t)(tokenBase + row) * DIM + col),
        (LDS_AS i32x4*)&xb_lds[row * XS + col], 0, 0);
  }
  __builtin_amdgcn_s_wait_asynccnt(0);
#else
  for (int i = 0; i < 16; ++i) {
    int c   = i * 256 + tid;        // 0..4095 chunks of 8 elems
    int row = c >> 7;
    int col = (c & 127) * 8;
    bf16x8 v = *(const bf16x8*)(xb + (size_t)(tokenBase + row) * DIM + col);
    *(bf16x8*)(&xb_lds[row * XS + col]) = v;
  }
#endif
  __syncthreads();

  // GEMM1 wave -> h tile (mt1, nt1) of the 32 x HC chunk
  const int mt1 = wave >> 2;        // 0..1
  const int nt1 = wave & 3;         // 0..3
  // GEMM2 wave -> 128-wide output column slab
  const int dslab = wave * 128;

  f32x8 acc[2][8];
  const f32x8 zf = {0.f, 0.f, 0.f, 0.f, 0.f, 0.f, 0.f, 0.f};
#pragma unroll
  for (int m = 0; m < 2; ++m)
#pragma unroll
    for (int n = 0; n < 8; ++n) acc[m][n] = zf;

  int pp = 0;  // ping-pong selector for h tile
  for (int e = 0; e < NE; ++e) {
    float gwl[8];                   // gate weight for my 8 accumulator rows (GEMM1 tile)
#pragma unroll
    for (int i = 0; i < 8; ++i) gwl[i] = wg_lds[mt1 * 16 + lhalf * 8 + i][e];

    for (int hc = 0; hc < HID / HC; ++hc) {
      const int hbase = hc * HC;

      // ---- GEMM1: h(mt1,nt1) = relu(x @ W1 + b1), K = D = 1024
      f32x8 hacc = zf;
      const int hglob = hbase + nt1 * 16 + lmod;
      const __bf16* aRow = &xb_lds[(mt1 * 16 + lmod) * XS + lhalf * 8];
      const size_t w1row = ((size_t)e * HID + hglob) * DIM + (size_t)(lhalf * 8);
      // per-lane prefetch of next chunk's W1 B-rows into WGP$
      if (hc + 1 < HID / HC)
        __builtin_prefetch((const void*)(w1t + w1row + (size_t)HC * DIM), 0, 1);
#pragma unroll 4
      for (int ks = 0; ks < DIM / 32; ++ks) {
        bf16x8 a0 = *(const bf16x8*)(aRow + ks * 32);
        bf16x8 a1 = *(const bf16x8*)(aRow + ks * 32 + 16);
        bf16x8 b0 = *(const bf16x8*)(w1t + w1row + ks * 32);
        bf16x8 b1v = *(const bf16x8*)(w1t + w1row + ks * 32 + 16);
        hacc = wmma_bf16(cat8(a0, a1), cat8(b0, b1v), hacc);
      }
      // bias + relu + gate-scale, native cvt to bf16, write h tile (ping-pong)
      const float b1s = b1g[(size_t)e * HID + hglob];
#pragma unroll
      for (int i = 0; i < 8; ++i) {
        float v = hacc[i] + b1s;
        v = v > 0.f ? v : 0.f;
        v *= gwl[i];
        int m = mt1 * 16 + lhalf * 8 + i;
        h_lds[pp][m * HS + nt1 * 16 + lmod] = (__bf16)v;
      }
      __syncthreads();

      // ---- GEMM2: acc += h @ W2, K = HC = 64 (gate scale already in h, so
      //      this accumulator sums over all H-chunks AND all experts)
#pragma unroll
      for (int ks2 = 0; ks2 < HC / 32; ++ks2) {
        const int kcol = ks2 * 32 + lhalf * 8;
        bf16x8 p0 = *(const bf16x8*)(&h_lds[pp][lmod * HS + kcol]);
        bf16x8 p1 = *(const bf16x8*)(&h_lds[pp][lmod * HS + kcol + 16]);
        bf16x8 q0 = *(const bf16x8*)(&h_lds[pp][(16 + lmod) * HS + kcol]);
        bf16x8 q1 = *(const bf16x8*)(&h_lds[pp][(16 + lmod) * HS + kcol + 16]);
        bf16x16 af0 = cat8(p0, p1);
        bf16x16 af1 = cat8(q0, q1);
#pragma unroll
        for (int nt = 0; nt < 8; ++nt) {
          const int dglob = dslab + nt * 16 + lmod;
          const size_t w2row =
              ((size_t)e * DIM + dglob) * HID + (size_t)(hbase + kcol);
          bf16x8 b0 = *(const bf16x8*)(w2t + w2row);
          bf16x8 b1q = *(const bf16x8*)(w2t + w2row + 16);
          bf16x16 bfr = cat8(b0, b1q);
          acc[0][nt] = wmma_bf16(af0, bfr, acc[0][nt]);
          acc[1][nt] = wmma_bf16(af1, bfr, acc[1][nt]);
        }
      }
      pp ^= 1;  // next GEMM1 writes the other buffer; no trailing barrier needed
    }
  }

  // epilogue: out = acc + sum_e gw[n,e]*b2[e,d]
#pragma unroll
  for (int nt = 0; nt < 8; ++nt) {
    const int dglob = dslab + nt * 16 + lmod;
    float b2v[NE];
#pragma unroll
    for (int e = 0; e < NE; ++e) b2v[e] = b2g[(size_t)e * DIM + dglob];
#pragma unroll
    for (int mt = 0; mt < 2; ++mt) {
#pragma unroll
      for (int i = 0; i < 8; ++i) {
        int m = mt * 16 + lhalf * 8 + i;
        float extra = 0.f;
#pragma unroll
        for (int e = 0; e < NE; ++e) extra += wg_lds[m][e] * b2v[e];
        out[(size_t)(tokenBase + m) * DIM + dglob] = acc[mt][nt][i] + extra;
      }
    }
  }
}

// ---------------------------------------------------------------- launcher
extern "C" void kernel_launch(void* const* d_in, const int* in_sizes, int n_in,
                              void* d_out, int out_size, void* d_ws, size_t ws_size,
                              hipStream_t stream) {
  const float* x  = (const float*)d_in[0];
  const float* Wg = (const float*)d_in[1];
  const float* bg = (const float*)d_in[2];
  const float* W1 = (const float*)d_in[3];
  const float* b1 = (const float*)d_in[4];
  const float* W2 = (const float*)d_in[5];
  const float* b2 = (const float*)d_in[6];
  float* out = (float*)d_out;

  char* ws = (char*)d_ws;
  const size_t XB = (size_t)N_TOK * DIM * 2;        //  16 MB x bf16
  const size_t WT = (size_t)NE * DIM * HID * 2;     //  64 MB each weight tensor
  __bf16* xb    = (__bf16*)(ws);
  __bf16* w1t   = (__bf16*)(ws + XB);
  __bf16* w2t   = (__bf16*)(ws + XB + WT);
  float*  gatew = (float*)(ws + XB + 2 * WT);

  // 1) convert x to bf16 (16 MB, L2-resident afterwards)
  cvt_x_kernel<<<(N_TOK * DIM) / 1024, 256, 0, stream>>>(x, xb);
  // 2) transpose+convert weights to bf16: 128 MB total -> fits 192 MB L2
  transpose_cvt_kernel<<<dim3(HID / 32, DIM / 32, NE), dim3(32, 8), 0, stream>>>(
      W1, w1t, DIM, HID);
  transpose_cvt_kernel<<<dim3(DIM / 32, HID / 32, NE), dim3(32, 8), 0, stream>>>(
      W2, w2t, HID, DIM);
  // 3) exact f32 softmax router
  gate_kernel<<<N_TOK, 256, 0, stream>>>(x, Wg, bg, gatew);
  // 4) fused GEMM1 -> relu -> gate-scale -> GEMM2 -> expert-combine
  moe_fused_kernel<<<N_TOK / TM, 256, 0, stream>>>(xb, w1t, w2t, gatew, b1, b2, out);
}
